// HadamardLinear_54073638256919
// MI455X (gfx1250) — matrix operand order repaired and checked by reading, not compiled, not run on hardware
//
#include <hip/hip_runtime.h>

typedef __attribute__((ext_vector_type(16))) _Float16 v16h;
typedef __attribute__((ext_vector_type(8)))  _Float16 v8h;
typedef __attribute__((ext_vector_type(8)))  float    v8f;

#define GROUPS   32
#define IN_O     128
#define OUT_O    128
#define D_DIM    4096
#define TOK_TILE 16
#define XM_ROW   136                    // 128 halfs + 8 pad (bank spread)
#define XM_TOK   (TOK_TILE * XM_ROW)    // 2176 halfs per group
#define W_ELEMS  (GROUPS * OUT_O * IN_O)  // 524288
#define RSQRT32  0.17677669529663687f

// ---------------- 32-point fast Walsh-Hadamard (Sylvester / natural order) ----
__device__ __forceinline__ void fwht32(float* v) {
#pragma unroll
  for (int len = 1; len < 32; len <<= 1) {
#pragma unroll
    for (int i = 0; i < 32; i += 2 * len) {
#pragma unroll
      for (int j = 0; j < len; ++j) {
        float a = v[i + j], b = v[i + j + len];
        v[i + j]       = a + b;
        v[i + j + len] = a - b;
      }
    }
  }
}

// ---------------- Pass 1a: per-block |w| partial sums (deterministic) --------
__global__ __launch_bounds__(256) void hl_abs_part(const float* __restrict__ w,
                                                   float* __restrict__ partial) {
  __shared__ float red[256];
  int base = blockIdx.x * 1024 + threadIdx.x;
  float s = 0.f;
#pragma unroll
  for (int r = 0; r < 4; ++r) s += fabsf(w[base + r * 256]);
  red[threadIdx.x] = s;
  __syncthreads();
  for (int off = 128; off > 0; off >>= 1) {
    if (threadIdx.x < off) red[threadIdx.x] += red[threadIdx.x + off];
    __syncthreads();
  }
  if (threadIdx.x == 0) partial[blockIdx.x] = red[0];
}

// ---------------- Pass 1b: final reduce -> scale = max(mean|w|, 1e-8) --------
__global__ __launch_bounds__(256) void hl_abs_final(const float* __restrict__ partial,
                                                    float* __restrict__ scale_out) {
  __shared__ float red[256];
  red[threadIdx.x] = partial[threadIdx.x] + partial[threadIdx.x + 256];
  __syncthreads();
  for (int off = 128; off > 0; off >>= 1) {
    if (threadIdx.x < off) red[threadIdx.x] += red[threadIdx.x + off];
    __syncthreads();
  }
  if (threadIdx.x == 0)
    scale_out[0] = fmaxf(red[0] / (float)W_ELEMS, 1e-8f);
}

// ---------------- Pass 2: ternary quantize -> f16 [g][o][i] ------------------
__global__ __launch_bounds__(256) void hl_quant(const float* __restrict__ w,
                                                const float* __restrict__ scale_p,
                                                _Float16* __restrict__ wq) {
  int idx = blockIdx.x * 256 + threadIdx.x;
  float s = scale_p[0];
  float q = fminf(fmaxf(rintf(w[idx] / s), -1.f), 1.f) * s;
  wq[idx] = (_Float16)q;
}

// ---------------- Main fused kernel: alpha*FWHT -> grouped WMMA -> FWHT*beta -
__global__ __launch_bounds__(256) void hl_main(const float* __restrict__ x,
                                               const _Float16* __restrict__ wq,
                                               const float* __restrict__ alpha,
                                               const float* __restrict__ beta,
                                               float* __restrict__ out) {
  extern __shared__ char smem[];
  _Float16* xm  = (_Float16*)smem;                                  // [32][16][136] f16
  float*    cbf = (float*)(smem + (size_t)GROUPS * XM_TOK * 2);     // [32][16][16]  f32

  const int tid  = threadIdx.x;
  const int lane = tid & 31;
  const int wave = tid >> 5;
  const int tok0 = blockIdx.x * TOK_TILE;

  // ---- Stage 1: x * alpha, FWHT over groups, f16 into LDS -------------------
  {
    const int i  = tid & 127;       // feature-in-group
    const int tb = tid >> 7;        // 0 or 1
    float av[GROUPS];
#pragma unroll
    for (int g = 0; g < GROUPS; ++g) av[g] = alpha[g * IN_O + i];
    for (int r = 0; r < 8; ++r) {
      const int t = tb + 2 * r;
      const float* xrow = x + (size_t)(tok0 + t) * D_DIM + i;
      float v[GROUPS];
#pragma unroll
      for (int g = 0; g < GROUPS; ++g) v[g] = xrow[g * IN_O] * av[g];
      fwht32(v);
#pragma unroll
      for (int g = 0; g < GROUPS; ++g)
        xm[g * XM_TOK + t * XM_ROW + i] = (_Float16)(v[g] * RSQRT32);
    }
  }
  __syncthreads();

  const int mn  = lane & 15;   // A-row (M) == B-col (N) index for this lane
  const int sub = lane >> 4;   // lane half selects contiguous 16-wide K sub-range

  for (int nc = 0; nc < 8; ++nc) {
    const int ob = nc * 16;    // output-feature chunk base

    // ---- Stage 2: grouped GEMM, 4 groups per wave, K=128 in 4 WMMA steps ----
#pragma unroll
    for (int j = 0; j < 4; ++j) {
      const int g = wave * 4 + j;
      v8f acc = {};
      const _Float16* brow  = wq + ((size_t)(g * OUT_O + ob + mn)) * IN_O;
      const _Float16* abase = xm + g * XM_TOK + mn * XM_ROW;
#pragma unroll
      for (int ks = 0; ks < 4; ++ks) {
        const int kb = ks * 32;
        union { v16h v; v8h h[2]; } a;
        a.h[0] = *(const v8h*)(abase + kb + sub * 16);       // K = kb+sub*16 .. +7
        a.h[1] = *(const v8h*)(abase + kb + sub * 16 + 8);   // K = kb+sub*16+8 .. +15
        v16h b = *(const v16h*)(brow + kb + sub * 16);       // K = kb+sub*16 .. +15
        acc = __builtin_amdgcn_wmma_f32_16x16x32_f16(
            false, a.v, false, b, (short)0, acc, false, false);
      }
      // D layout: VGPR r -> M = r + 8*sub, N = mn
#pragma unroll
      for (int r = 0; r < 8; ++r)
        cbf[g * 256 + (r + 8 * sub) * 16 + mn] = acc[r];
    }
    __syncthreads();

    // ---- Stage 3: FWHT over groups, * beta, store ---------------------------
    {
      const int t = tid >> 4, o = tid & 15;
      float v[GROUPS];
#pragma unroll
      for (int g = 0; g < GROUPS; ++g) v[g] = cbf[g * 256 + t * 16 + o];
      fwht32(v);
      float* orow = out + (size_t)(tok0 + t) * D_DIM + ob + o;
#pragma unroll
      for (int h = 0; h < GROUPS; ++h)
        orow[h * OUT_O] = v[h] * RSQRT32 * beta[h * OUT_O + ob + o];
    }
    __syncthreads();   // cbf reused next chunk
  }
}

extern "C" void kernel_launch(void* const* d_in, const int* in_sizes, int n_in,
                              void* d_out, int out_size, void* d_ws, size_t ws_size,
                              hipStream_t stream) {
  const float* x      = (const float*)d_in[0];
  const float* weight = (const float*)d_in[1];
  const float* alpha  = (const float*)d_in[2];
  const float* beta   = (const float*)d_in[3];
  float* out = (float*)d_out;

  // Workspace layout: [0,4): scale | [256, 256+2048): 512 partials | [4096, +1MB): wq f16
  float*    scale   = (float*)d_ws;
  float*    partial = (float*)((char*)d_ws + 256);
  _Float16* wqf16   = (_Float16*)((char*)d_ws + 4096);

  hl_abs_part <<<512, 256, 0, stream>>>(weight, partial);
  hl_abs_final<<<1,   256, 0, stream>>>(partial, scale);
  hl_quant    <<<W_ELEMS / 256, 256, 0, stream>>>(weight, scale, wqf16);

  const int tokens = 4 * 2048;
  const size_t lds = (size_t)GROUPS * XM_TOK * 2 + (size_t)GROUPS * 256 * 4; // 172032 B
  hl_main<<<tokens / TOK_TILE, 256, lds, stream>>>(x, wqf16, alpha, beta, out);
}
